// SelfAttention_74878459838554
// MI455X (gfx1250) — matrix operand order repaired and checked by reading, not compiled, not run on hardware
//
#include <hip/hip_runtime.h>
#include <hip/hip_bf16.h>

typedef __attribute__((ext_vector_type(16))) __bf16 v16bf;
typedef __attribute__((ext_vector_type(8)))  __bf16 v8bf;
typedef __attribute__((ext_vector_type(8)))  float  v8f;

#define BB 2
#define TT 2048
#define CC 2048
#define HH 16
#define HD 128

__device__ inline v8f zero8() {
    v8f z = {0.f, 0.f, 0.f, 0.f, 0.f, 0.f, 0.f, 0.f};
    return z;
}

__device__ inline v8f wmma_bf16(v16bf a, v16bf b, v8f c) {
    return __builtin_amdgcn_wmma_f32_16x16x32_bf16(
        false, a, false, b, (short)0, c, false, false);
}

// ---- CDNA5 async global->LDS copy (ASYNCcnt-tracked), 16 bytes per lane ----
__device__ __forceinline__ void async_copy_b128(unsigned lds_off, const void* gaddr) {
    asm volatile("global_load_async_to_lds_b128 %0, %1, off"
                 :: "v"(lds_off), "v"(gaddr) : "memory");
}
__device__ __forceinline__ void wait_async0() {
    asm volatile("s_wait_asynccnt 0" ::: "memory");
}
__device__ __forceinline__ unsigned lds_addr(const void* p) {
    // generic LDS pointer = {shared aperture, 32-bit LDS byte address}
    return (unsigned)(uintptr_t)p;
}

// A-matrix fragment (16x32 bf16), source row-major [m][k] in LDS.
// lane<16 -> M=lane, K=e / 16+(e-8); lane>=16 -> K base +8.
__device__ inline v16bf load_A_frag(const __bf16* base, int stride) {
    const int lane = threadIdx.x & 31;
    const __bf16* p = base + (lane & 15) * stride + ((lane >> 4) << 3);
    v8bf lo = *(const v8bf*)(p);
    v8bf hi = *(const v8bf*)(p + 16);
    v16bf r;
#pragma unroll
    for (int e = 0; e < 8; ++e) { r[e] = lo[e]; r[e + 8] = hi[e]; }
    return r;
}

// B-matrix fragment (32x16 bf16), source (n,k) row-major in LDS.
// N = lane&15, K = (lane>>4)*8 + e / 16+(lane>>4)*8+(e-8).
__device__ inline v16bf load_B_frag(const __bf16* base, int stride) {
    const int lane = threadIdx.x & 31;
    const __bf16* p = base + (lane & 15) * stride + ((lane >> 4) << 3);
    v8bf lo = *(const v8bf*)(p);
    v8bf hi = *(const v8bf*)(p + 16);
    v16bf r;
#pragma unroll
    for (int e = 0; e < 8; ++e) { r[e] = lo[e]; r[e + 8] = hi[e]; }
    return r;
}

// ---------------------------------------------------------------------------
// fp32 -> bf16 bulk convert (8 elems/thread)
// ---------------------------------------------------------------------------
__global__ __launch_bounds__(256) void cvt_f32_bf16(
    const float* __restrict__ in, __bf16* __restrict__ out)
{
    const size_t i = ((size_t)blockIdx.x * 256 + threadIdx.x) * 8;
    const float4 a = *(const float4*)(in + i);
    const float4 b = *(const float4*)(in + i + 4);
    v8bf o;
    o[0] = (__bf16)a.x; o[1] = (__bf16)a.y; o[2] = (__bf16)a.z; o[3] = (__bf16)a.w;
    o[4] = (__bf16)b.x; o[5] = (__bf16)b.y; o[6] = (__bf16)b.z; o[7] = (__bf16)b.w;
    *(v8bf*)(out + i) = o;
}

// ---------------------------------------------------------------------------
// GEMM: C[m][n] = sum_k A[m][k] * W[n][k]  (A @ W^T), A/W bf16, C fp32.
// BM=64, BN=64, BK=64, double-buffered LDS fed by async b128 copies.
// 256 threads = 8 waves; wave (wm,wn) owns 16x32 of C -> 4 WMMA / K-stage.
// ---------------------------------------------------------------------------
__global__ __launch_bounds__(256) void gemm_bf16_async(
    const __bf16* __restrict__ A, int lda,
    const __bf16* __restrict__ W, int ldw,
    float* __restrict__ Cout, int ldc, int K)
{
    __shared__ __bf16 As[2][64 * 72];   // stride 72 elems = 144B (16B multiple)
    __shared__ __bf16 Ws[2][64 * 72];

    const int tid  = threadIdx.x;
    const int wave = tid >> 5;
    const int lane = tid & 31;
    const int wm = wave & 3;
    const int wn = wave >> 2;
    const int m0 = blockIdx.y * 64;
    const int n0 = blockIdx.x * 64;

    // 64x64 bf16 tile = 512 x 16B chunks; thread covers chunks tid, tid+256
    const int r0 = tid >> 3,         kc0 = (tid & 7) * 8;
    const int r1 = (tid + 256) >> 3, kc1 = ((tid + 256) & 7) * 8;

    const __bf16* Arow0 = A + (size_t)(m0 + r0) * lda + kc0;
    const __bf16* Arow1 = A + (size_t)(m0 + r1) * lda + kc1;
    const __bf16* Wrow0 = W + (size_t)(n0 + r0) * ldw + kc0;
    const __bf16* Wrow1 = W + (size_t)(n0 + r1) * ldw + kc1;

    // prologue: stage tile 0
    async_copy_b128(lds_addr(&As[0][r0 * 72 + kc0]), Arow0);
    async_copy_b128(lds_addr(&As[0][r1 * 72 + kc1]), Arow1);
    async_copy_b128(lds_addr(&Ws[0][r0 * 72 + kc0]), Wrow0);
    async_copy_b128(lds_addr(&Ws[0][r1 * 72 + kc1]), Wrow1);
    wait_async0();
    __syncthreads();

    v8f acc0 = zero8();
    v8f acc1 = zero8();
    const int niter = K >> 6;

    for (int it = 0; it < niter; ++it) {
        const int buf = it & 1;
        if (it + 1 < niter) {
            const int nb = buf ^ 1;
            const int ko = (it + 1) << 6;
            async_copy_b128(lds_addr(&As[nb][r0 * 72 + kc0]), Arow0 + ko);
            async_copy_b128(lds_addr(&As[nb][r1 * 72 + kc1]), Arow1 + ko);
            async_copy_b128(lds_addr(&Ws[nb][r0 * 72 + kc0]), Wrow0 + ko);
            async_copy_b128(lds_addr(&Ws[nb][r1 * 72 + kc1]), Wrow1 + ko);
        }
        const __bf16* as = &As[buf][(wm * 16) * 72];
        const __bf16* ws = &Ws[buf][(wn * 32) * 72];
#pragma unroll
        for (int kk = 0; kk < 2; ++kk) {
            v16bf a  = load_A_frag(as + kk * 32, 72);
            v16bf b0 = load_B_frag(ws + kk * 32, 72);
            v16bf b1 = load_B_frag(ws + 16 * 72 + kk * 32, 72);
            acc0 = wmma_bf16(a, b0, acc0);
            acc1 = wmma_bf16(a, b1, acc1);
        }
        wait_async0();
        __syncthreads();
    }

    const int nl = lane & 15;
    const int mb = (lane >> 4) * 8;
    float* cp = Cout + (size_t)(m0 + wm * 16 + mb) * ldc + n0 + wn * 32;
#pragma unroll
    for (int v = 0; v < 8; ++v) {
        cp[(size_t)v * ldc + nl]      = acc0[v];
        cp[(size_t)v * ldc + 16 + nl] = acc1[v];
    }
}

// ---------------------------------------------------------------------------
// RoPE + scatter. Reads qkv fp32 (B,T,3C). Writes:
//   q_r  bf16 (B,H,T,hd)  rope'd, pre-scaled by 1/sqrt(hd)
//   k_r  bf16 (B,H,T,hd)  rope'd
//   v_t  bf16 (B,H,hd,T)  transposed for PV B-fragments
//   k_out fp32 (B,T,H,hd) pre-rope   [output 2]
//   v_out fp32 (B,H,T,hd)            [output 3]
// ---------------------------------------------------------------------------
__global__ __launch_bounds__(256) void rope_scatter(
    const float* __restrict__ qkv,
    __bf16* __restrict__ q_r, __bf16* __restrict__ k_r, __bf16* __restrict__ v_t,
    float* __restrict__ k_out, float* __restrict__ v_out)
{
    const int idx = blockIdx.x * 256 + threadIdx.x;   // B*T*H*64 total
    const int j = idx & 63;
    const int h = (idx >> 6) & (HH - 1);
    const int t = (idx >> 10) & (TT - 1);
    const int b = idx >> 21;

    const float theta = __expf(-(float)j * (9.210340371976184f / 64.0f));
    const float ang = (float)t * theta;
    float s, c;
    sincosf(ang, &s, &c);
    const float scale = 0.08838834764831845f;   // 1/sqrt(128)

    const size_t base = ((size_t)(b * TT + t)) * (3 * CC) + h * HD + 2 * j;
    const float q0 = qkv[base],          q1 = qkv[base + 1];
    const float k0 = qkv[base + CC],     k1 = qkv[base + CC + 1];
    const float v0 = qkv[base + 2 * CC], v1 = qkv[base + 2 * CC + 1];

    const size_t bhtd = (((size_t)(b * HH + h)) * TT + t) * HD + 2 * j;
    q_r[bhtd]     = (__bf16)((q0 * c - q1 * s) * scale);
    q_r[bhtd + 1] = (__bf16)((q1 * c + q0 * s) * scale);
    k_r[bhtd]     = (__bf16)(k0 * c - k1 * s);
    k_r[bhtd + 1] = (__bf16)(k1 * c + k0 * s);
    v_out[bhtd]     = v0;
    v_out[bhtd + 1] = v1;

    const size_t bhdt = (((size_t)(b * HH + h)) * HD + 2 * j) * TT + t;
    v_t[bhdt]      = (__bf16)v0;
    v_t[bhdt + TT] = (__bf16)v1;

    const size_t bthd = (((size_t)(b * TT + t)) * HH + h) * HD + 2 * j;
    k_out[bthd]     = k0;
    k_out[bthd + 1] = k1;
}

// ---------------------------------------------------------------------------
// Flash attention (non-causal). 128 threads = 4 waves; 64 q rows per block.
// K/V tiles (64 keys) double-buffered in LDS via async b128 copies.
// ---------------------------------------------------------------------------
__global__ __launch_bounds__(128) void flash_attn(
    const __bf16* __restrict__ q_r, const __bf16* __restrict__ k_r,
    const __bf16* __restrict__ v_t, __bf16* __restrict__ o_bf)
{
    __shared__ __bf16 Qs[64 * 136];        // (q,d)
    __shared__ __bf16 Ks[2][64 * 136];     // (key,d)  = (n,k)
    __shared__ __bf16 Vt[2][128 * 72];     // (d,key)  = (n,k)
    __shared__ __bf16 Ps[4 * 16 * 72];     // per-wave P (q,key)

    const int tid  = threadIdx.x;
    const int wave = tid >> 5;
    const int lane = tid & 31;
    const int bh  = blockIdx.y;            // b*16 + h
    const int q0g = blockIdx.x * 64;

    const __bf16* qb = q_r + ((size_t)bh * TT + q0g) * HD;
    const __bf16* kb = k_r + (size_t)bh * TT * HD;
    const __bf16* vb = v_t + (size_t)bh * HD * TT;

    // stage Q (64x128 bf16 = 1024 chunks) + K/V tile 0
#pragma unroll
    for (int i = 0; i < 8; ++i) {
        const int c = tid + i * 128;
        const int r = c >> 4, kc = (c & 15) * 8;           // 16 chunks / row
        async_copy_b128(lds_addr(&Qs[r * 136 + kc]), qb + (size_t)r * HD + kc);
    }
#pragma unroll
    for (int i = 0; i < 8; ++i) {
        const int c = tid + i * 128;
        const int r = c >> 4, kc = (c & 15) * 8;
        async_copy_b128(lds_addr(&Ks[0][r * 136 + kc]), kb + (size_t)r * HD + kc);
    }
#pragma unroll
    for (int i = 0; i < 8; ++i) {
        const int c = tid + i * 128;
        const int r = c >> 3, kc = (c & 7) * 8;            // 8 chunks / row (64 keys)
        async_copy_b128(lds_addr(&Vt[0][r * 72 + kc]), vb + (size_t)r * TT + kc);
    }
    wait_async0();
    __syncthreads();

    v16bf qa[4];
#pragma unroll
    for (int kk = 0; kk < 4; ++kk)
        qa[kk] = load_A_frag(&Qs[(wave * 16) * 136 + kk * 32], 136);

    float mrow[8], lrow[8];
    v8f oacc[8];
#pragma unroll
    for (int v = 0; v < 8; ++v) { mrow[v] = -3.0e38f; lrow[v] = 0.f; }
#pragma unroll
    for (int n = 0; n < 8; ++n) oacc[n] = zero8();

    __bf16* pbuf = &Ps[wave * 16 * 72];
    const int nl = lane & 15;
    const int mb = (lane >> 4) * 8;

    for (int kt = 0; kt < TT / 64; ++kt) {
        const int buf = kt & 1;
        if (kt + 1 < TT / 64) {
            const int nb = buf ^ 1;
            const __bf16* kp = kb + (size_t)(kt + 1) * 64 * HD;
            const __bf16* vp = vb + (size_t)(kt + 1) * 64;
#pragma unroll
            for (int i = 0; i < 8; ++i) {
                const int c = tid + i * 128;
                const int r = c >> 4, kc = (c & 15) * 8;
                async_copy_b128(lds_addr(&Ks[nb][r * 136 + kc]), kp + (size_t)r * HD + kc);
            }
#pragma unroll
            for (int i = 0; i < 8; ++i) {
                const int c = tid + i * 128;
                const int r = c >> 3, kc = (c & 7) * 8;
                async_copy_b128(lds_addr(&Vt[nb][r * 72 + kc]), vp + (size_t)r * TT + kc);
            }
        }

        // S = Q @ K^T : 16 x 64 per wave
        v8f sacc[4];
#pragma unroll
        for (int nt = 0; nt < 4; ++nt) {
            v8f acc = zero8();
#pragma unroll
            for (int kk = 0; kk < 4; ++kk) {
                v16bf bf = load_B_frag(&Ks[buf][(nt * 16) * 136 + kk * 32], 136);
                acc = wmma_bf16(qa[kk], bf, acc);
            }
            sacc[nt] = acc;
        }

        // --- online softmax (fp32) ---
        float mnew[8];
#pragma unroll
        for (int v = 0; v < 8; ++v) mnew[v] = mrow[v];
#pragma unroll
        for (int nt = 0; nt < 4; ++nt)
#pragma unroll
            for (int v = 0; v < 8; ++v) mnew[v] = fmaxf(mnew[v], sacc[nt][v]);
        for (int off = 1; off < 16; off <<= 1)
#pragma unroll
            for (int v = 0; v < 8; ++v)
                mnew[v] = fmaxf(mnew[v], __shfl_xor(mnew[v], off, 32));

        float alpha[8], rsum[8];
#pragma unroll
        for (int v = 0; v < 8; ++v) {
            alpha[v] = __expf(mrow[v] - mnew[v]);
            mrow[v] = mnew[v];
            rsum[v] = 0.f;
        }
#pragma unroll
        for (int nt = 0; nt < 4; ++nt)
#pragma unroll
            for (int v = 0; v < 8; ++v) {
                const float p = __expf(sacc[nt][v] - mnew[v]);
                sacc[nt][v] = p;
                rsum[v] += p;
            }
        for (int off = 1; off < 16; off <<= 1)
#pragma unroll
            for (int v = 0; v < 8; ++v)
                rsum[v] += __shfl_xor(rsum[v], off, 32);
#pragma unroll
        for (int v = 0; v < 8; ++v) lrow[v] = lrow[v] * alpha[v] + rsum[v];
#pragma unroll
        for (int n8 = 0; n8 < 8; ++n8)
#pragma unroll
            for (int v = 0; v < 8; ++v) oacc[n8][v] = oacc[n8][v] * alpha[v];

        // P -> bf16 LDS (q,key) for A-fragment reload (wave-private region)
#pragma unroll
        for (int nt = 0; nt < 4; ++nt)
#pragma unroll
            for (int v = 0; v < 8; ++v)
                pbuf[(mb + v) * 72 + nt * 16 + nl] = (__bf16)sacc[nt][v];

        v16bf pa0 = load_A_frag(&pbuf[0], 72);
        v16bf pa1 = load_A_frag(&pbuf[32], 72);
#pragma unroll
        for (int n8 = 0; n8 < 8; ++n8) {
            v8f acc = oacc[n8];
            v16bf vb0 = load_B_frag(&Vt[buf][(n8 * 16) * 72], 72);
            v16bf vb1 = load_B_frag(&Vt[buf][(n8 * 16) * 72 + 32], 72);
            acc = wmma_bf16(pa0, vb0, acc);
            acc = wmma_bf16(pa1, vb1, acc);
            oacc[n8] = acc;
        }

        wait_async0();
        __syncthreads();
    }

    // epilogue: o /= l -> bf16 (B,T,C) scratch feeding proj GEMM
    const int b = bh >> 4, h = bh & 15;
    __bf16* op = o_bf + ((size_t)(b * TT) + (q0g + wave * 16)) * CC + h * HD;
#pragma unroll
    for (int n8 = 0; n8 < 8; ++n8)
#pragma unroll
        for (int v = 0; v < 8; ++v)
            op[(size_t)(mb + v) * CC + n8 * 16 + nl] = (__bf16)(oacc[n8][v] / lrow[v]);
}

// ---------------------------------------------------------------------------
extern "C" void kernel_launch(void* const* d_in, const int* in_sizes, int n_in,
                              void* d_out, int out_size, void* d_ws, size_t ws_size,
                              hipStream_t stream) {
    const float* x      = (const float*)d_in[0];
    const float* w_qkv  = (const float*)d_in[1];
    const float* w_proj = (const float*)d_in[2];

    float* y_out = (float*)d_out;                       // (B,T,C)
    float* k_out = y_out + (size_t)BB * TT * CC;        // (B,T,H,hd)
    float* v_out = k_out + (size_t)BB * TT * CC;        // (B,H,T,hd)

    const size_t NX  = (size_t)BB * TT * CC;            // 8,388,608
    const size_t NWQ = (size_t)3 * CC * CC;             // 12,582,912
    const size_t NWP = (size_t)CC * CC;                 // 4,194,304

    __bf16* x_bf  = (__bf16*)d_ws;
    __bf16* wq_bf = x_bf + NX;
    __bf16* wp_bf = wq_bf + NWQ;
    float*  qkv   = (float*)(wp_bf + NWP);              // (B*T, 3C) fp32
    __bf16* q_r   = (__bf16*)(qkv + (size_t)BB * TT * 3 * CC);
    __bf16* k_r   = q_r + NX;
    __bf16* v_t   = k_r + NX;
    __bf16* o_bf  = x_bf;                               // reuse after GEMM1

    // 0) convert inputs to bf16
    cvt_f32_bf16<<<dim3(NX  / 2048), 256, 0, stream>>>(x,      x_bf);
    cvt_f32_bf16<<<dim3(NWQ / 2048), 256, 0, stream>>>(w_qkv,  wq_bf);
    cvt_f32_bf16<<<dim3(NWP / 2048), 256, 0, stream>>>(w_proj, wp_bf);

    // 1) qkv = x @ w_qkv^T   (4096 x 6144, K=2048)
    gemm_bf16_async<<<dim3(3 * CC / 64, BB * TT / 64), 256, 0, stream>>>(
        x_bf, CC, wq_bf, CC, qkv, 3 * CC, CC);

    // 2) RoPE + scatter outputs / attention operands
    rope_scatter<<<dim3((BB * TT * HH * 64) / 256), 256, 0, stream>>>(
        qkv, q_r, k_r, v_t, k_out, v_out);

    // 3) flash attention -> o_bf (B,T,C)
    flash_attn<<<dim3(TT / 64, BB * HH), 128, 0, stream>>>(q_r, k_r, v_t, o_bf);

    // 4) y = o @ w_proj^T    (4096 x 2048, K=2048)
    gemm_bf16_async<<<dim3(CC / 64, BB * TT / 64), 256, 0, stream>>>(
        o_bf, CC, wp_bf, CC, y_out, CC, CC);
}
